// CGCNN_61246233641659
// MI455X (gfx1250) — matrix operand order repaired and checked by reading, not compiled, not run on hardware
//
#include <hip/hip_runtime.h>

typedef _Float16 v16h __attribute__((ext_vector_type(16)));
typedef _Float16 h8   __attribute__((ext_vector_type(8)));
typedef float    v8f  __attribute__((ext_vector_type(8)));

constexpr int N_NODES  = 50000;
constexpr int N_EDGES  = 1600000;
constexpr int NODE_F   = 92;
constexpr int EDGE_F   = 41;
constexpr int HID      = 128;
constexpr int N_CONV   = 3;
constexpr int N_GRAPHS = 256;
constexpr int ZDIM     = 2 * HID + EDGE_F;   // 297
constexpr int ZPAD     = 320;                // pad K to multiple of 32
constexpr int KSTEPS   = ZPAD / 32;          // 10
constexpr int NTILES   = HID / 16;           // 8
constexpr int TILE_M   = 32;                 // edges per block
constexpr long FRAG_HALFS = (long)NTILES * KSTEPS * 32 * 16; // 40960 per matrix

__device__ __forceinline__ float sigmoidf_(float x) {
    return 1.0f / (1.0f + __expf(-x));
}
__device__ __forceinline__ float softplusf_(float x) {
    return (x > 20.0f) ? x : log1pf(__expf(x));
}

// ---------------------------------------------------------------------------
// Pre-pack msg_w / gate_w (f32 [3][297][128]) into per-lane WMMA B fragments,
// f16, K padded to 320.  Layout: [layer][mat][ntile 8][kstep 10][lane 32][16].
// B fragment, this code's contiguous K order: lanes 0-15 hold K=kk..kk+15 of column n,
// lanes 16-31 hold K=kk+16..kk+31.
// ---------------------------------------------------------------------------
__global__ __launch_bounds__(256) void pack_weights_k(
    const float* __restrict__ msg_w, const float* __restrict__ gate_w,
    _Float16* __restrict__ wpack)
{
    long idx = (long)blockIdx.x * 256 + threadIdx.x;
    const long TOT = (long)N_CONV * 2 * FRAG_HALFS;
    if (idx >= TOT) return;
    long o = idx;
    int i     = (int)(o & 15); o >>= 4;
    int lane  = (int)(o & 31); o >>= 5;
    int step  = (int)(o % KSTEPS); o /= KSTEPS;
    int ntile = (int)(o & 7);  o >>= 3;
    int mat   = (int)(o & 1);  o >>= 1;
    int l     = (int)o;
    int n = ntile * 16 + (lane & 15);
    int k = step * 32 + (lane >> 4) * 16 + i;
    const float* src = (mat == 0) ? msg_w : gate_w;
    float v = (k < ZDIM) ? src[((long)l * ZDIM + k) * HID + n] : 0.0f;
    wpack[idx] = (_Float16)v;
}

// ---------------------------------------------------------------------------
// Node embedding: h = x @ emb_w + emb_b    [50000,92]x[92,128]
// ---------------------------------------------------------------------------
__global__ __launch_bounds__(256) void embed_k(
    const float* __restrict__ x, const float* __restrict__ w,
    const float* __restrict__ b, float* __restrict__ h)
{
    long i = (long)blockIdx.x * 256 + threadIdx.x;
    if (i >= (long)N_NODES * HID) return;
    int node = (int)(i >> 7);
    int c    = (int)(i & 127);
    const float* xr = x + (long)node * NODE_F;
    float acc = b[c];
#pragma unroll 4
    for (int k = 0; k < NODE_F; ++k)
        acc = fmaf(xr[k], w[(long)k * HID + c], acc);
    h[i] = acc;
}

// ---------------------------------------------------------------------------
// Edge conv (WMMA core).  Block = 256 threads = 8 waves; 32-edge tile.
// z = [x_i | x_j | edge_attr | 0-pad] staged in LDS as f16 (32 x 320).
// Wave w computes columns [16w,16w+16) of gate=sigmoid(z@wm+bm) and
// msg=softplus(z@wg+bg) for both 16-row M-tiles; contrib scattered with
// f32 atomics into aggr[dst].
// ---------------------------------------------------------------------------
__global__ __launch_bounds__(256) void edge_conv_k(
    const float* __restrict__ h,        // [N_NODES,128]
    const int*   __restrict__ ei,       // [2,E]
    const float* __restrict__ ea,       // [E,41]
    const v16h*  __restrict__ wm_pack,  // packed fragments (gate weights)
    const v16h*  __restrict__ wg_pack,  // packed fragments (msg weights)
    const float* __restrict__ bm,       // [128]
    const float* __restrict__ bg,       // [128]
    float* __restrict__ aggr)           // [N_NODES,128]
{
    __shared__ _Float16 zt[TILE_M][ZPAD] __attribute__((aligned(16)));
    __shared__ int dstt[TILE_M];
    __shared__ int srct[TILE_M];

    const int t  = threadIdx.x;
    const long e0 = (long)blockIdx.x * TILE_M;

    if (t < TILE_M) {
        dstt[t] = ei[e0 + t];
        srct[t] = ei[(long)N_EDGES + e0 + t];
    }
    __syncthreads();

    // --- stage x_i (k 0..127) and x_j (k 128..255) as f16 -------------------
#pragma unroll
    for (int p = 0; p < 2; ++p) {
        int m = (p * 256 + t) >> 4;        // 0..31
        int c = (t & 15) * 8;              // 0,8,...,120
        const float4* pi = (const float4*)(h + (long)dstt[m] * HID + c);
        const float4* pj = (const float4*)(h + (long)srct[m] * HID + c);
        float4 a0 = pi[0], a1 = pi[1];
        float4 b0 = pj[0], b1 = pj[1];
        h8 vi, vj;
        vi[0] = (_Float16)a0.x; vi[1] = (_Float16)a0.y; vi[2] = (_Float16)a0.z; vi[3] = (_Float16)a0.w;
        vi[4] = (_Float16)a1.x; vi[5] = (_Float16)a1.y; vi[6] = (_Float16)a1.z; vi[7] = (_Float16)a1.w;
        vj[0] = (_Float16)b0.x; vj[1] = (_Float16)b0.y; vj[2] = (_Float16)b0.z; vj[3] = (_Float16)b0.w;
        vj[4] = (_Float16)b1.x; vj[5] = (_Float16)b1.y; vj[6] = (_Float16)b1.z; vj[7] = (_Float16)b1.w;
        *(h8*)&zt[m][c]       = vi;
        *(h8*)&zt[m][HID + c] = vj;
    }
    // --- stage edge_attr (k 256..296) ---------------------------------------
    for (int idx = t; idx < TILE_M * EDGE_F; idx += 256) {
        int m = idx / EDGE_F;
        int k = idx - m * EDGE_F;
        zt[m][2 * HID + k] = (_Float16)ea[(e0 + m) * EDGE_F + k];
    }
    // --- zero pad (k 297..319) ----------------------------------------------
    for (int idx = t; idx < TILE_M * 32; idx += 256) {
        int m = idx >> 5;
        int k = 288 + (idx & 31);
        if (k >= ZDIM) zt[m][k] = (_Float16)0.0f;
    }
    __syncthreads();

    const int wave    = t >> 5;
    const int lane    = t & 31;
    const int lane_lo = lane & 15;
    const int hi      = lane >> 4;      // half-wave
    const int ncol    = wave * 16 + lane_lo;

    const v16h* wmf = wm_pack + (long)(wave * KSTEPS) * 32 + lane;
    const v16h* wgf = wg_pack + (long)(wave * KSTEPS) * 32 + lane;

    v8f cg0 = {}, cm0 = {}, cg1 = {}, cm1 = {};

#pragma unroll
    for (int s = 0; s < KSTEPS; ++s) {
        const int kk = s * 32;
        // A fragments (contiguous K order matching B: halves 0-15 = K kk+16*hi..kk+16*hi+15)
        h8 a0lo = *(const h8*)&zt[lane_lo][kk + 16 * hi];
        h8 a0hi = *(const h8*)&zt[lane_lo][kk + 16 * hi + 8];
        h8 a1lo = *(const h8*)&zt[16 + lane_lo][kk + 16 * hi];
        h8 a1hi = *(const h8*)&zt[16 + lane_lo][kk + 16 * hi + 8];
        v16h a0, a1;
#pragma unroll
        for (int i = 0; i < 8; ++i) {
            a0[i] = a0lo[i]; a0[8 + i] = a0hi[i];
            a1[i] = a1lo[i]; a1[8 + i] = a1hi[i];
        }
        v16h bfm = wmf[s * 32];   // gate-branch weights
        v16h bfg = wgf[s * 32];   // msg-branch weights

        cg0 = __builtin_amdgcn_wmma_f32_16x16x32_f16(false, a0, false, bfm, (short)0, cg0, false, false);
        cm0 = __builtin_amdgcn_wmma_f32_16x16x32_f16(false, a0, false, bfg, (short)0, cm0, false, false);
        cg1 = __builtin_amdgcn_wmma_f32_16x16x32_f16(false, a1, false, bfm, (short)0, cg1, false, false);
        cm1 = __builtin_amdgcn_wmma_f32_16x16x32_f16(false, a1, false, bfg, (short)0, cm1, false, false);
    }

    const float bmv = bm[ncol];
    const float bgv = bg[ncol];
#pragma unroll
    for (int r = 0; r < 8; ++r) {
        // C/D layout: VGPR r, lanes<16 -> M=r, lanes>=16 -> M=r+8
        int m0 = r + 8 * hi;
        float v0 = sigmoidf_(cg0[r] + bmv) * softplusf_(cm0[r] + bgv);
        atomicAdd(&aggr[(long)dstt[m0] * HID + ncol], v0);
        int m1 = 16 + r + 8 * hi;
        float v1 = sigmoidf_(cg1[r] + bmv) * softplusf_(cm1[r] + bgv);
        atomicAdd(&aggr[(long)dstt[m1] * HID + ncol], v1);
    }
}

// ---------------------------------------------------------------------------
// h_out = softplus(h_in + aggr)
// ---------------------------------------------------------------------------
__global__ __launch_bounds__(256) void node_update_k(
    const float* __restrict__ hin, const float* __restrict__ aggr,
    float* __restrict__ hout)
{
    long i = (long)blockIdx.x * 256 + threadIdx.x;
    if (i >= (long)N_NODES * HID) return;
    hout[i] = softplusf_(hin[i] + aggr[i]);
}

// ---------------------------------------------------------------------------
// Graph pooling: atomic scatter of h into per-graph sums + counts
// ---------------------------------------------------------------------------
__global__ __launch_bounds__(256) void pool_scatter_k(
    const float* __restrict__ h, const int* __restrict__ batch,
    float* __restrict__ pooled, float* __restrict__ cnt)
{
    long i = (long)blockIdx.x * 256 + threadIdx.x;
    if (i >= (long)N_NODES * HID) return;
    int node = (int)(i >> 7);
    int c    = (int)(i & 127);
    int g    = batch[node];
    atomicAdd(&pooled[(long)g * HID + c], h[i]);
    if (c == 0) atomicAdd(&cnt[g], 1.0f);
}

// ---------------------------------------------------------------------------
// Readout: out[g] = softplus(mean_g @ ro1 + b1) @ ro2 + b2
// One block (128 threads) per graph.
// ---------------------------------------------------------------------------
__global__ __launch_bounds__(128) void readout_k(
    const float* __restrict__ pooled, const float* __restrict__ cnt,
    const float* __restrict__ w1, const float* __restrict__ b1,
    const float* __restrict__ w2, const float* __restrict__ b2,
    float* __restrict__ out)
{
    __shared__ float prow[HID];
    __shared__ float tbuf[HID];
    int g = blockIdx.x;
    int n = threadIdx.x;
    float c = fmaxf(cnt[g], 1.0f);
    prow[n] = pooled[(long)g * HID + n] / c;
    __syncthreads();
    float acc = b1[n];
#pragma unroll 4
    for (int k = 0; k < HID; ++k)
        acc = fmaf(prow[k], w1[(long)k * HID + n], acc);
    tbuf[n] = softplusf_(acc) * w2[n];
    __syncthreads();
    for (int s = 64; s > 0; s >>= 1) {
        if (n < s) tbuf[n] += tbuf[n + s];
        __syncthreads();
    }
    if (n == 0) out[g] = tbuf[0] + b2[0];
}

// ---------------------------------------------------------------------------
extern "C" void kernel_launch(void* const* d_in, const int* in_sizes, int n_in,
                              void* d_out, int out_size, void* d_ws, size_t ws_size,
                              hipStream_t stream)
{
    (void)in_sizes; (void)n_in; (void)out_size; (void)ws_size;

    const float* x      = (const float*)d_in[0];
    const int*   ei     = (const int*)  d_in[1];
    const float* ea     = (const float*)d_in[2];
    const int*   batch  = (const int*)  d_in[3];
    const float* emb_w  = (const float*)d_in[4];
    const float* emb_b  = (const float*)d_in[5];
    const float* msg_w  = (const float*)d_in[6];
    const float* msg_b  = (const float*)d_in[7];
    const float* gate_w = (const float*)d_in[8];
    const float* gate_b = (const float*)d_in[9];
    const float* ro1_w  = (const float*)d_in[10];
    const float* ro1_b  = (const float*)d_in[11];
    const float* ro2_w  = (const float*)d_in[12];
    const float* ro2_b  = (const float*)d_in[13];
    float* out = (float*)d_out;

    char* ws = (char*)d_ws;
    size_t off = 0;
    auto carve = [&](size_t bytes) -> void* {
        void* p = ws + off;
        off = (off + bytes + 255) & ~(size_t)255;
        return p;
    };
    const size_t HBYTES = (size_t)N_NODES * HID * sizeof(float);
    float*    h_a    = (float*)carve(HBYTES);
    float*    h_b    = (float*)carve(HBYTES);
    float*    aggr   = (float*)carve(HBYTES);
    _Float16* wpack  = (_Float16*)carve((size_t)N_CONV * 2 * FRAG_HALFS * sizeof(_Float16));
    float*    pooled = (float*)carve((size_t)N_GRAPHS * HID * sizeof(float));
    float*    cnt    = (float*)carve((size_t)N_GRAPHS * sizeof(float));

    // 1) pack weights to f16 WMMA fragments
    {
        long tot = (long)N_CONV * 2 * FRAG_HALFS;
        pack_weights_k<<<(int)((tot + 255) / 256), 256, 0, stream>>>(msg_w, gate_w, wpack);
    }
    // 2) embedding
    {
        long tot = (long)N_NODES * HID;
        embed_k<<<(int)((tot + 255) / 256), 256, 0, stream>>>(x, emb_w, emb_b, h_a);
    }
    // 3) conv layers
    float* cur = h_a;
    float* nxt = h_b;
    for (int l = 0; l < N_CONV; ++l) {
        hipMemsetAsync(aggr, 0, HBYTES, stream);
        const v16h* wm_l = (const v16h*)(wpack + (long)(l * 2 + 0) * FRAG_HALFS);
        const v16h* wg_l = (const v16h*)(wpack + (long)(l * 2 + 1) * FRAG_HALFS);
        edge_conv_k<<<N_EDGES / TILE_M, 256, 0, stream>>>(
            cur, ei, ea, wm_l, wg_l, msg_b + (long)l * HID, gate_b + (long)l * HID, aggr);
        long tot = (long)N_NODES * HID;
        node_update_k<<<(int)((tot + 255) / 256), 256, 0, stream>>>(cur, aggr, nxt);
        float* tmp = cur; cur = nxt; nxt = tmp;
    }
    // 4) pooling + readout
    hipMemsetAsync(pooled, 0, (size_t)N_GRAPHS * HID * sizeof(float), stream);
    hipMemsetAsync(cnt, 0, (size_t)N_GRAPHS * sizeof(float), stream);
    {
        long tot = (long)N_NODES * HID;
        pool_scatter_k<<<(int)((tot + 255) / 256), 256, 0, stream>>>(cur, batch, pooled, cnt);
    }
    readout_k<<<N_GRAPHS, HID, 0, stream>>>(pooled, cnt, ro1_w, ro1_b, ro2_w, ro2_b, out);
}
